// FrustumSegmentationNet_49529562858053
// MI455X (gfx1250) — compile-verified, matrix operand order checked by reading, not run on hardware
//
#include <hip/hip_runtime.h>
#include <hip/hip_bf16.h>
#include <float.h>

// ---------------- problem constants ----------------
#define NPTS   65536      // 256*256 points (full box)
#define NSAMP  500        // FPS samples
#define KNB    50         // ball-query neighbors
#define RADIUS 0.001f
#define BN_EPS 1e-5f
#define RG     (NSAMP * KNB)   // 25000 grouped rows
#define RGP    25024           // RG padded to 64 (GEMM block rows)
#define NSP    512             // NSAMP padded to 64

typedef __attribute__((ext_vector_type(16))) _Float16 v16h;
typedef __attribute__((ext_vector_type(8)))  _Float16 v8h;
typedef __attribute__((ext_vector_type(8)))  float    v8f;

// =====================================================================
// 1) Build point cloud: pc[r][0:3]=coords, [3:6]=rgb
// =====================================================================
__global__ void build_pc_kernel(const float* __restrict__ rgb,
                                const float* __restrict__ depth,
                                const float* __restrict__ intr,
                                float* __restrict__ pc) {
    int r = blockIdx.x * blockDim.x + threadIdx.x;
    if (r >= NPTS) return;
    int v = r >> 8, u = r & 255;
    float fx = intr[0], cx = intr[2], fy = intr[4], cy = intr[5];
    float z = depth[r];
    pc[r * 6 + 0] = (u + 0.5f - cx) / fx * z;
    pc[r * 6 + 1] = (v + 0.5f - cy) / fy * z;
    pc[r * 6 + 2] = z;
    pc[r * 6 + 3] = rgb[r * 3 + 0];
    pc[r * 6 + 4] = rgb[r * 3 + 1];
    pc[r * 6 + 5] = rgb[r * 3 + 2];
}

// =====================================================================
// 2) Farthest point sampling: single persistent workgroup, LDS argmin
// =====================================================================
__global__ __launch_bounds__(1024) void fps_kernel(const float* __restrict__ pc,
                                                   float* __restrict__ dist,
                                                   int* __restrict__ sel) {
    __shared__ float sV[1024];
    __shared__ int   sI[1024];
    __shared__ int   sBest;
    const int t = threadIdx.x;
    const float x0 = pc[0], y0 = pc[1], z0 = pc[2];
    for (int i = t; i < NPTS; i += 1024) {
        float dx = pc[i * 6 + 0] - x0, dy = pc[i * 6 + 1] - y0, dz = pc[i * 6 + 2] - z0;
        dist[i] = dx * dx + dy * dy + dz * dz;
    }
    if (t == 0) sel[0] = 0;
    __syncthreads();
    for (int s = 1; s < NSAMP; ++s) {
        float bv = FLT_MAX; int bi = 0x7fffffff;
        for (int i = t; i < NPTS; i += 1024) {
            float v = dist[i];
            if (v < bv || (v == bv && i < bi)) { bv = v; bi = i; }
        }
        sV[t] = bv; sI[t] = bi;
        __syncthreads();
        for (int st = 512; st > 0; st >>= 1) {
            if (t < st) {
                float ov = sV[t + st]; int oi = sI[t + st];
                if (ov < sV[t] || (ov == sV[t] && oi < sI[t])) { sV[t] = ov; sI[t] = oi; }
            }
            __syncthreads();
        }
        if (t == 0) { sBest = sI[0]; sel[s] = sI[0]; }
        __syncthreads();
        const int b = sBest;
        const float bx = pc[b * 6], by = pc[b * 6 + 1], bz = pc[b * 6 + 2];
        for (int i = t; i < NPTS; i += 1024) {
            float dx = pc[i * 6] - bx, dy = pc[i * 6 + 1] - by, dz = pc[i * 6 + 2] - bz;
            float nd = dx * dx + dy * dy + dz * dz;
            if (nd < dist[i]) dist[i] = nd;
        }
        __syncthreads();
    }
}

// =====================================================================
// 3) Ball query: 256 KB distance array resident in CDNA5's 320 KB LDS
// =====================================================================
__global__ void ball_query_kernel(const float* __restrict__ pc,
                                  const int* __restrict__ sel,
                                  int* __restrict__ idxK,
                                  int* __restrict__ maskArr,
                                  float* __restrict__ grouped) {
    extern __shared__ float sm[];
    float* ld = sm;                    // NPTS floats (256 KB)
    float* sV = sm + NPTS;             // 256 floats
    int*   sI = (int*)(sV + 256);      // 256 ints
    const int g = blockIdx.x, t = threadIdx.x;
    const int si = sel[g];
    const float sx = pc[si * 6], sy = pc[si * 6 + 1], sz = pc[si * 6 + 2];
    int cnt = 0;
    for (int i = t; i < NPTS; i += 256) {
        float dx = pc[i * 6] - sx, dy = pc[i * 6 + 1] - sy, dz = pc[i * 6 + 2] - sz;
        float d = sqrtf(dx * dx + dy * dy + dz * dz);
        ld[i] = d;
        if (d < RADIUS) cnt++;
    }
    sI[t] = cnt;
    __syncthreads();
    for (int st = 128; st > 0; st >>= 1) {
        if (t < st) sI[t] += sI[t + st];
        __syncthreads();
    }
    if (t == 0) { int m = sI[0]; maskArr[g] = (m > KNB) ? KNB : m; }
    __syncthreads();
    for (int j = 0; j < KNB; ++j) {
        float bv = FLT_MAX; int bi = 0x7fffffff;
        for (int i = t; i < NPTS; i += 256) {
            float v = ld[i];
            if (v < bv || (v == bv && i < bi)) { bv = v; bi = i; }
        }
        sV[t] = bv; sI[t] = bi;
        __syncthreads();
        for (int st = 128; st > 0; st >>= 1) {
            if (t < st) {
                float ov = sV[t + st]; int oi = sI[t + st];
                if (ov < sV[t] || (ov == sV[t] && oi < sI[t])) { sV[t] = ov; sI[t] = oi; }
            }
            __syncthreads();
        }
        if (t == 0) {
            int b = sI[0]; float v = sV[0];
            idxK[g * KNB + j] = b;
            ld[b] = FLT_MAX;
            float f = (v < RADIUS) ? 1.f : 0.f;
            for (int d = 0; d < 6; ++d)
                grouped[(g * KNB + j) * 6 + d] = pc[b * 6 + d] * f;
        }
        __syncthreads();
    }
}

// =====================================================================
// 4) Per-group centering + scaling, f32 grouped -> padded f16 h0
// =====================================================================
__global__ void center_kernel(const float* __restrict__ grouped,
                              _Float16* __restrict__ h0h) {
    __shared__ float mean[3], mx[3], mn[3], mx2[3], mn2[3];
    const int g = blockIdx.x, t = threadIdx.x;
    const float* grp = grouped + (size_t)g * KNB * 6;
    if (t == 0) {
        for (int d = 0; d < 3; ++d) { mean[d] = 0.f; mx[d] = -FLT_MAX; mn[d] = FLT_MAX; }
        for (int j = 0; j < KNB; ++j)
            for (int d = 0; d < 3; ++d) {
                float v = grp[j * 6 + d];
                mean[d] += v;
                mx[d] = fmaxf(mx[d], v);
                mn[d] = fminf(mn[d], v);
            }
        for (int d = 0; d < 3; ++d) {
            mean[d] /= KNB;
            float len = mx[d] - mn[d];
            mx2[d] = (mx[d] - mean[d]) / len;
            mn2[d] = (mn[d] - mean[d]) / len;
        }
    }
    __syncthreads();
    for (int j = t; j < KNB; j += blockDim.x) {
        _Float16* orow = h0h + (size_t)(g * KNB + j) * 32;
        for (int d = 0; d < 3; ++d) {
            float len = mx[d] - mn[d];
            float c = (grp[j * 6 + d] - mean[d]) / len;
            float len2 = (mx2[d] - mn2[d]) * 0.5f;
            orow[d] = (_Float16)(c * 10.f / len2);
        }
        orow[3] = (_Float16)grp[j * 6 + 3];
        orow[4] = (_Float16)grp[j * 6 + 4];
        orow[5] = (_Float16)grp[j * 6 + 5];
        // cols 6..31 already zero from the bulk fill
    }
}

// =====================================================================
// 5) WMMA GEMM, branch-free + register-blocked:
//    each wave: 1 A fragment x 4 B fragments -> 4 accumulators
//    (16 rows x 64 cols per wave; block of 4 waves = 64x64 tile).
//    Load:wmma ratio = 2.5 b128 per v_wmma (vs 4 for 1x1 blocking).
// =====================================================================
__global__ __launch_bounds__(128) void gemm_wmma_kernel(
        const _Float16* __restrict__ A, int lda,
        const _Float16* __restrict__ B, int ldb,
        const float* __restrict__ bias, float* __restrict__ C, int ldc,
        int nchunks) {
    const int lane = threadIdx.x & 31;
    const int wv   = threadIdx.x >> 5;
    const int m0   = blockIdx.y * 64 + wv * 16;   // wave's 16 rows
    const int n0   = blockIdx.x * 64;             // wave's 64 cols
    const int half = lane >> 4;
    const int mr   = lane & 15;
    const _Float16* Ap = A + (size_t)(m0 + mr) * lda + half * 8;
    const _Float16* Bp = B + (size_t)(n0 + mr) * ldb + half * 8;
    const size_t bstride = (size_t)16 * ldb;      // 16 output-channel rows

    v8f acc0 = {}, acc1 = {}, acc2 = {}, acc3 = {};
    #pragma unroll 2
    for (int ck = 0; ck < nchunks; ++ck) {
        v8h a0 = *(const v8h*)Ap;
        v8h a1 = *(const v8h*)(Ap + 16);
        v8h b00 = *(const v8h*)(Bp);
        v8h b01 = *(const v8h*)(Bp + 16);
        v8h b10 = *(const v8h*)(Bp + bstride);
        v8h b11 = *(const v8h*)(Bp + bstride + 16);
        v8h b20 = *(const v8h*)(Bp + 2 * bstride);
        v8h b21 = *(const v8h*)(Bp + 2 * bstride + 16);
        v8h b30 = *(const v8h*)(Bp + 3 * bstride);
        v8h b31 = *(const v8h*)(Bp + 3 * bstride + 16);
        Ap += 32; Bp += 32;
        v16h a, b0, b1, b2, b3;
        #pragma unroll
        for (int i = 0; i < 8; ++i) {
            a[i] = a0[i];  a[8 + i] = a1[i];
            b0[i] = b00[i]; b0[8 + i] = b01[i];
            b1[i] = b10[i]; b1[8 + i] = b11[i];
            b2[i] = b20[i]; b2[8 + i] = b21[i];
            b3[i] = b30[i]; b3[8 + i] = b31[i];
        }
        acc0 = __builtin_amdgcn_wmma_f32_16x16x32_f16(false, a, false, b0, (short)0, acc0, false, false);
        acc1 = __builtin_amdgcn_wmma_f32_16x16x32_f16(false, a, false, b1, (short)0, acc1, false, false);
        acc2 = __builtin_amdgcn_wmma_f32_16x16x32_f16(false, a, false, b2, (short)0, acc2, false, false);
        acc3 = __builtin_amdgcn_wmma_f32_16x16x32_f16(false, a, false, b3, (short)0, acc3, false, false);
    }
    // epilogue: D layout -> row = m0 + r + 8*half, col = n0 + t*16 + mr
    v8f accs[4] = {acc0, acc1, acc2, acc3};
    #pragma unroll
    for (int tl = 0; tl < 4; ++tl) {
        const int col = n0 + tl * 16 + mr;
        const float bb = bias[col];
        #pragma unroll
        for (int r = 0; r < 8; ++r)
            C[(size_t)(m0 + r + 8 * half) * ldc + col] = accs[tl][r] + bb;
    }
}

// =====================================================================
// 6) weight/bias packing: f32 (N,K) -> zero-padded f16 (Ncols,Kp) + bias pad
// =====================================================================
__global__ void pack_w_kernel(const float* __restrict__ W, const float* __restrict__ b,
                              _Float16* __restrict__ Wh, float* __restrict__ bp,
                              int N, int K, int Kp, int Ncols) {
    size_t idx = (size_t)blockIdx.x * blockDim.x + threadIdx.x;
    size_t tot = (size_t)Ncols * Kp;
    if (idx >= tot) return;
    int n = (int)(idx / Kp), k = (int)(idx % Kp);
    Wh[idx] = (n < N && k < K) ? (_Float16)W[(size_t)n * K + k] : (_Float16)0.f;
    if (k == 0) bp[n] = (n < N) ? b[n] : 0.f;
}

// =====================================================================
// 7) BatchNorm stats + apply (f32 in, padded f16 out); C is a power of 2
// =====================================================================
__global__ void bn_stats_kernel(const float* __restrict__ X, int ld, int R, int C,
                                float* __restrict__ mean, float* __restrict__ invstd) {
    __shared__ float s1[256], s2[256];
    const int c = blockIdx.x, t = threadIdx.x;
    float a = 0.f, b = 0.f;
    for (int r = t; r < R; r += 256) {
        float v = X[(size_t)r * ld + c];
        a += v; b += v * v;
    }
    s1[t] = a; s2[t] = b;
    __syncthreads();
    for (int st = 128; st > 0; st >>= 1) {
        if (t < st) { s1[t] += s1[t + st]; s2[t] += s2[t + st]; }
        __syncthreads();
    }
    if (t == 0) {
        float m = s1[0] / (float)R;
        float v = s2[0] / (float)R - m * m;
        mean[c] = m;
        invstd[c] = rsqrtf(v + BN_EPS);
    }
}

__global__ void bn_apply_kernel(const float* __restrict__ X, int R,
                                int cshift,            // C = 1 << cshift; ld == C
                                const float* __restrict__ mean,
                                const float* __restrict__ invstd,
                                const float* __restrict__ gamma,
                                const float* __restrict__ beta, int relu,
                                _Float16* __restrict__ Y) {
    size_t idx = (size_t)blockIdx.x * blockDim.x + threadIdx.x;
    size_t tot = (size_t)R << cshift;
    if (idx >= tot) return;
    int c = (int)(idx & ((1u << cshift) - 1u));
    float y = (X[idx] - mean[c]) * invstd[c] * gamma[c] + beta[c];
    if (relu) y = fmaxf(y, 0.f);
    Y[idx] = (_Float16)y;
}

// =====================================================================
// 8) Max-pool over K=50 neighbors (f16 in/out, C=1024)
// =====================================================================
__global__ void maxpool_kernel(const _Float16* __restrict__ X, _Float16* __restrict__ P) {
    const int g = blockIdx.x;
    for (int c = threadIdx.x; c < 1024; c += blockDim.x) {
        float m = -FLT_MAX;
        for (int j = 0; j < KNB; ++j)
            m = fmaxf(m, (float)X[((size_t)g * KNB + j) * 1024 + c]);
        P[(size_t)g * 1024 + c] = (_Float16)m;
    }
}

// =====================================================================
// 9) scatter-max feature index + concat + output transpose
// =====================================================================
__global__ void fill_zero_u32(unsigned int* p, size_t n) {
    size_t i = (size_t)blockIdx.x * blockDim.x + threadIdx.x;
    if (i < n) p[i] = 0u;
}

__global__ void scatter_find_kernel(const int* __restrict__ idxK,
                                    const int* __restrict__ maskArr,
                                    int* __restrict__ find) {
    const int g = blockIdx.x, j = threadIdx.x;
    if (j < KNB && j < maskArr[g])
        atomicMax(&find[idxK[g * KNB + j]], g);
}

__global__ void build_xcat_kernel(const float* __restrict__ pc,
                                  const float* __restrict__ featc,   // ld 128
                                  const int* __restrict__ find,
                                  _Float16* __restrict__ xcath) {    // ld 160
    const int r = blockIdx.x, t = threadIdx.x;
    const int fi = find[r];
    if (t < 6) xcath[(size_t)r * 160 + t] = (_Float16)pc[r * 6 + t];
    for (int j = t; j < 128; j += blockDim.x)
        xcath[(size_t)r * 160 + 6 + j] = (_Float16)featc[(size_t)fi * 128 + j];
}

__global__ void write_out_kernel(const float* __restrict__ segc, float* __restrict__ out) {
    size_t idx = (size_t)blockIdx.x * blockDim.x + threadIdx.x;
    if (idx >= (size_t)80 * NPTS) return;
    int c = (int)(idx / NPTS);
    int r = (int)(idx % NPTS);
    out[idx] = segc[(size_t)r * 128 + c];
}

// =====================================================================
// Host-side launch
// =====================================================================
static inline void launch_gemm(const _Float16* A, int lda, const _Float16* B, int ldb,
                               const float* bias, float* C, int ldc,
                               int Mp, int Ncols, int Kp, hipStream_t s) {
    dim3 grid(Ncols / 64, Mp / 64);
    gemm_wmma_kernel<<<grid, 128, 0, s>>>(A, lda, B, ldb, bias, C, ldc, Kp / 32);
}

static inline void launch_pack(const float* W, const float* b, _Float16* Wh, float* bp,
                               int N, int K, int Kp, int Ncols, hipStream_t s) {
    size_t tot = (size_t)Ncols * Kp;
    pack_w_kernel<<<(unsigned)((tot + 255) / 256), 256, 0, s>>>(W, b, Wh, bp, N, K, Kp, Ncols);
}

extern "C" void kernel_launch(void* const* d_in, const int* in_sizes, int n_in,
                              void* d_out, int out_size, void* d_ws, size_t ws_size,
                              hipStream_t stream) {
    const float* rgb   = (const float*)d_in[0];
    const float* depth = (const float*)d_in[1];
    const float* intr  = (const float*)d_in[2];
    const float *W1 = (const float*)d_in[4],  *b1 = (const float*)d_in[5];
    const float *g1 = (const float*)d_in[6],  *be1 = (const float*)d_in[7];
    const float *W2 = (const float*)d_in[8],  *b2 = (const float*)d_in[9];
    const float *g2 = (const float*)d_in[10], *be2 = (const float*)d_in[11];
    const float *W3 = (const float*)d_in[12], *b3 = (const float*)d_in[13];
    const float *g3 = (const float*)d_in[14], *be3 = (const float*)d_in[15];
    const float *Wd1 = (const float*)d_in[16], *bd1 = (const float*)d_in[17];
    const float *g4 = (const float*)d_in[18], *be4 = (const float*)d_in[19];
    const float *Wd2 = (const float*)d_in[20], *bd2 = (const float*)d_in[21];
    const float *g5 = (const float*)d_in[22], *be5 = (const float*)d_in[23];
    const float *Wd3 = (const float*)d_in[24], *bd3 = (const float*)d_in[25];
    const float *Ws = (const float*)d_in[26], *bs = (const float*)d_in[27];

    char* w = (char*)d_ws;
    auto alloc = [&](size_t bytes) -> void* {
        void* p = (void*)w;
        w += (bytes + 255) & ~(size_t)255;
        return p;
    };
    // ---- f32 region ----
    float* pc      = (float*)alloc((size_t)NPTS * 6 * 4);
    float* fpsDist = (float*)alloc((size_t)NPTS * 4);
    int*   fpsIdx  = (int*)alloc((size_t)NSAMP * 4);
    int*   idxK    = (int*)alloc((size_t)NSAMP * KNB * 4);
    int*   maskArr = (int*)alloc((size_t)NSAMP * 4);
    float* grouped = (float*)alloc((size_t)RG * 6 * 4);
    float* h1c     = (float*)alloc((size_t)RGP * 64 * 4);
    float* h2c     = (float*)alloc((size_t)RGP * 128 * 4);
    float* h3c     = (float*)alloc((size_t)RGP * 1024 * 4);
    float* d1c     = (float*)alloc((size_t)NSP * 512 * 4);
    float* d2c     = (float*)alloc((size_t)NSP * 256 * 4);
    float* featc   = (float*)alloc((size_t)NSP * 128 * 4);
    float* segc    = (float*)alloc((size_t)NPTS * 128 * 4);
    float* mean    = (float*)alloc(1024 * 4);
    float* invstd  = (float*)alloc(1024 * 4);
    int*   find    = (int*)alloc((size_t)NPTS * 4);
    float* bp1 = (float*)alloc(64 * 4),  *bp2 = (float*)alloc(128 * 4);
    float* bp3 = (float*)alloc(1024 * 4), *bpd1 = (float*)alloc(512 * 4);
    float* bpd2 = (float*)alloc(256 * 4), *bpd3 = (float*)alloc(128 * 4);
    float* bps = (float*)alloc(128 * 4);
    // ---- f16 region (zeroed in one shot) ----
    char* f16_begin = w;
    _Float16* h0h   = (_Float16*)alloc((size_t)RGP * 32 * 2);
    _Float16* h1h   = (_Float16*)alloc((size_t)RGP * 64 * 2);
    _Float16* h2h   = (_Float16*)alloc((size_t)RGP * 128 * 2);
    _Float16* h3h   = (_Float16*)alloc((size_t)RGP * 1024 * 2);
    _Float16* poolh = (_Float16*)alloc((size_t)NSP * 1024 * 2);
    _Float16* dd1h  = (_Float16*)alloc((size_t)NSP * 512 * 2);
    _Float16* dd2h  = (_Float16*)alloc((size_t)NSP * 256 * 2);
    _Float16* xcath = (_Float16*)alloc((size_t)NPTS * 160 * 2);
    _Float16* W1h  = (_Float16*)alloc((size_t)64 * 32 * 2);
    _Float16* W2h  = (_Float16*)alloc((size_t)128 * 64 * 2);
    _Float16* W3h  = (_Float16*)alloc((size_t)1024 * 128 * 2);
    _Float16* Wd1h = (_Float16*)alloc((size_t)512 * 1024 * 2);
    _Float16* Wd2h = (_Float16*)alloc((size_t)256 * 512 * 2);
    _Float16* Wd3h = (_Float16*)alloc((size_t)128 * 256 * 2);
    _Float16* Wsh  = (_Float16*)alloc((size_t)128 * 160 * 2);
    char* f16_end = w;

    // zero whole f16 region (padding must be 0 for branch-free WMMA GEMM)
    size_t zwords = (size_t)(f16_end - f16_begin) / 4;
    fill_zero_u32<<<(unsigned)((zwords + 255) / 256), 256, 0, stream>>>((unsigned*)f16_begin, zwords);
    // zero scatter target
    fill_zero_u32<<<(NPTS + 255) / 256, 256, 0, stream>>>((unsigned*)find, NPTS);

    // pack weights + biases (padded, f16)
    launch_pack(W1, b1, W1h, bp1, 64, 6, 32, 64, stream);
    launch_pack(W2, b2, W2h, bp2, 128, 64, 64, 128, stream);
    launch_pack(W3, b3, W3h, bp3, 1024, 128, 128, 1024, stream);
    launch_pack(Wd1, bd1, Wd1h, bpd1, 512, 1024, 1024, 512, stream);
    launch_pack(Wd2, bd2, Wd2h, bpd2, 256, 512, 512, 256, stream);
    launch_pack(Wd3, bd3, Wd3h, bpd3, 128, 256, 256, 128, stream);
    launch_pack(Ws, bs, Wsh, bps, 80, 134, 160, 128, stream);

    // 1) point cloud
    build_pc_kernel<<<NPTS / 256, 256, 0, stream>>>(rgb, depth, intr, pc);
    // 2) FPS
    fps_kernel<<<1, 1024, 0, stream>>>(pc, fpsDist, fpsIdx);
    // 3) ball query (distance array resident in 320 KB LDS)
    size_t bq_lds = (size_t)NPTS * 4 + 256 * 4 + 256 * 4;
    ball_query_kernel<<<NSAMP, 256, bq_lds, stream>>>(pc, fpsIdx, idxK, maskArr, grouped);
    // 4) center + rescale -> padded f16 h0
    center_kernel<<<NSAMP, 64, 0, stream>>>(grouped, h0h);

    // 5-11) PointNet trunk (all GEMMs branch-free register-blocked WMMA)
    launch_gemm(h0h, 32, W1h, 32, bp1, h1c, 64, RGP, 64, 32, stream);
    bn_stats_kernel<<<64, 256, 0, stream>>>(h1c, 64, RG, 64, mean, invstd);
    bn_apply_kernel<<<(unsigned)(((size_t)RG * 64 + 255) / 256), 256, 0, stream>>>(
        h1c, RG, 6, mean, invstd, g1, be1, 1, h1h);
    launch_gemm(h1h, 64, W2h, 64, bp2, h2c, 128, RGP, 128, 64, stream);
    bn_stats_kernel<<<128, 256, 0, stream>>>(h2c, 128, RG, 128, mean, invstd);
    bn_apply_kernel<<<(unsigned)(((size_t)RG * 128 + 255) / 256), 256, 0, stream>>>(
        h2c, RG, 7, mean, invstd, g2, be2, 1, h2h);
    launch_gemm(h2h, 128, W3h, 128, bp3, h3c, 1024, RGP, 1024, 128, stream);
    bn_stats_kernel<<<1024, 256, 0, stream>>>(h3c, 1024, RG, 1024, mean, invstd);
    bn_apply_kernel<<<(unsigned)(((size_t)RG * 1024 + 255) / 256), 256, 0, stream>>>(
        h3c, RG, 10, mean, invstd, g3, be3, 0, h3h);
    maxpool_kernel<<<NSAMP, 256, 0, stream>>>(h3h, poolh);
    launch_gemm(poolh, 1024, Wd1h, 1024, bpd1, d1c, 512, NSP, 512, 1024, stream);
    bn_stats_kernel<<<512, 256, 0, stream>>>(d1c, 512, NSAMP, 512, mean, invstd);
    bn_apply_kernel<<<(unsigned)(((size_t)NSAMP * 512 + 255) / 256), 256, 0, stream>>>(
        d1c, NSAMP, 9, mean, invstd, g4, be4, 1, dd1h);
    launch_gemm(dd1h, 512, Wd2h, 512, bpd2, d2c, 256, NSP, 256, 512, stream);
    bn_stats_kernel<<<256, 256, 0, stream>>>(d2c, 256, NSAMP, 256, mean, invstd);
    bn_apply_kernel<<<(unsigned)(((size_t)NSAMP * 256 + 255) / 256), 256, 0, stream>>>(
        d2c, NSAMP, 8, mean, invstd, g5, be5, 1, dd2h);
    launch_gemm(dd2h, 256, Wd3h, 256, bpd3, featc, 128, NSP, 128, 256, stream);

    // 17) scatter feature index
    scatter_find_kernel<<<NSAMP, 64, 0, stream>>>(idxK, maskArr, find);
    // 18) concat pc + gathered feats -> padded f16
    build_xcat_kernel<<<NPTS, 128, 0, stream>>>(pc, featc, find, xcath);
    // 19) segmentation head
    launch_gemm(xcath, 160, Wsh, 160, bps, segc, 128, NPTS, 128, 160, stream);
    // 20) (N,80) -> (80,H,W)
    size_t tot = (size_t)80 * NPTS;
    write_out_kernel<<<(unsigned)((tot + 255) / 256), 256, 0, stream>>>(segc, (float*)d_out);
}
